// MultiHeadAttention_40690520162931
// MI455X (gfx1250) — matrix operand order repaired and checked by reading, not compiled, not run on hardware
//
#include <hip/hip_runtime.h>
#include <hip/hip_bf16.h>
#include <stdint.h>

// Problem constants (reference: D_MODEL=1024, NUM_HEADS=16, HEAD_DIM=64, B=2, S=2048)
#define D_MODEL 1024
#define SEQ     2048
#define BATCH   2
#define NHEAD   16
#define DHEAD   64
#define MTOT    (BATCH * SEQ)   // 4096 rows

typedef __attribute__((ext_vector_type(16))) __bf16 v16bf;
typedef __attribute__((ext_vector_type(8)))  float  v8f;

union Frag16 {
    v16bf v;
    uint4 q[2];
    unsigned short h[16];
};

// fp32 -> bf16, round-to-nearest-even
static __device__ __forceinline__ unsigned short f2bf(float f) {
    unsigned u = __float_as_uint(f);
    unsigned r = u + 0x7FFFu + ((u >> 16) & 1u);
    return (unsigned short)(r >> 16);
}

static __device__ __forceinline__ v8f v8f_splat(float x) {
    v8f r;
#pragma unroll
    for (int i = 0; i < 8; ++i) r[i] = x;
    return r;
}

// ---------------------------------------------------------------------------
// Cross-lane xor-max over the 16-lane halves of a wave32.
// v_permlane16_b32 (VALU) instead of ds_bpermute (LDS pipe) when available.
// ---------------------------------------------------------------------------
#if __has_builtin(__builtin_amdgcn_permlane16)
static __device__ __forceinline__ float perm16(float x, unsigned lo, unsigned hi) {
    unsigned u = __float_as_uint(x);
    u = __builtin_amdgcn_permlane16(u, u, lo, hi, false, false);
    return __uint_as_float(u);
}
static __device__ __forceinline__ float xmax16(float x) {
    x = fmaxf(x, perm16(x, 0x67452301u, 0xEFCDAB89u));   // xor 1
    x = fmaxf(x, perm16(x, 0x54761032u, 0xDCFE98BAu));   // xor 2
    x = fmaxf(x, perm16(x, 0x32107654u, 0xBA98FEDCu));   // xor 4
    x = fmaxf(x, perm16(x, 0xFEDCBA98u, 0x76543210u));   // xor 8
    return x;
}
#else
static __device__ __forceinline__ float xmax16(float x) {
    x = fmaxf(x, __shfl_xor(x, 1, 32));
    x = fmaxf(x, __shfl_xor(x, 2, 32));
    x = fmaxf(x, __shfl_xor(x, 4, 32));
    x = fmaxf(x, __shfl_xor(x, 8, 32));
    return x;
}
#endif

// ---------------------------------------------------------------------------
// C[M,N] = oscale * (A[M,1024] @ W[N,1024]^T)   (torch Linear semantics)
// MODE 0: write bf16 into [B,H,S,Dh]  (Q/K projections; Q carries log2e/sqrt(Dh))
// MODE 1: write fp32 row-major [M,N]  (output projection -> d_out)
// MODE 2: write bf16 into [B,H,Dh,S'] (V projection, transposed + key-permuted:
//         within each 64-key block, position j holds key 16*(j&3)+(j>>2))
// Block: 128 threads (4 waves), computes a 64x64 output tile.
// ---------------------------------------------------------------------------
template <int MODE>
__global__ __launch_bounds__(128) void gemm_xWT(
    const float* __restrict__ A,
    const float* __restrict__ W,
    unsigned short* __restrict__ outBF,
    float* __restrict__ outF,
    float oscale)
{
    // Row stride 48 ushorts = 96 B -> every row 16B-aligned for b128 LDS ops.
    __shared__ __align__(16) unsigned short As[64][48];
    __shared__ __align__(16) unsigned short Ws[64][48];

    const int tid  = threadIdx.x;
    const int wid  = tid >> 5;        // wave id 0..3 (wave32)
    const int lane = tid & 31;
    const int l    = lane & 15;
    const int hi   = lane >> 4;       // 0: lanes 0-15, 1: lanes 16-31

    const int n0 = blockIdx.x * 64;
    const int m0 = blockIdx.y * 64;

    v8f acc[4];
#pragma unroll
    for (int t = 0; t < 4; ++t) acc[t] = v8f_splat(0.0f);

    for (int kk = 0; kk < D_MODEL; kk += 32) {
        // Stage 64x32 fp32 tiles of A and W, convert to bf16 in LDS.
#pragma unroll
        for (int i = 0; i < 4; ++i) {
            int idx = i * 128 + tid;      // 0..511 chunks of 4 floats
            int r   = idx >> 3;           // 0..63
            int c   = (idx & 7) * 4;      // 0..28
            float4 av = *(const float4*)&A[(size_t)(m0 + r) * D_MODEL + kk + c];
            float4 wv = *(const float4*)&W[(size_t)(n0 + r) * D_MODEL + kk + c];
            uint2 ap, wp;
            ap.x = (unsigned)f2bf(av.x) | ((unsigned)f2bf(av.y) << 16);
            ap.y = (unsigned)f2bf(av.z) | ((unsigned)f2bf(av.w) << 16);
            wp.x = (unsigned)f2bf(wv.x) | ((unsigned)f2bf(wv.y) << 16);
            wp.y = (unsigned)f2bf(wv.z) | ((unsigned)f2bf(wv.w) << 16);
            *(uint2*)&As[r][c] = ap;
            *(uint2*)&Ws[r][c] = wp;
        }
        __syncthreads();

        // A fragment (16x32): lanes<16 hold K {0..7,16..23}; lanes>=16 {8..15,24..31}
        Frag16 a;
        const int k0 = hi ? 8 : 0;
        a.q[0] = *(const uint4*)&As[wid * 16 + l][k0];
        a.q[1] = *(const uint4*)&As[wid * 16 + l][k0 + 16];

        const int kb = hi ? 16 : 0;   // B frag uses same K split as A (k0, k0+16)
#pragma unroll
        for (int t = 0; t < 4; ++t) {
            Frag16 b;   // column n = t*16 + l ; B(k,n) = W[n][k]
            b.q[0] = *(const uint4*)&Ws[t * 16 + l][k0];
            b.q[1] = *(const uint4*)&Ws[t * 16 + l][k0 + 16];
            acc[t] = __builtin_amdgcn_wmma_f32_16x16x32_bf16(
                false, a.v, false, b.v, (short)0, acc[t], false, false);
        }
        __syncthreads();
    }

    // C fragment layout: col = l, row = hi*8 + v
#pragma unroll
    for (int t = 0; t < 4; ++t) {
#pragma unroll
        for (int v = 0; v < 8; ++v) {
            int m = m0 + wid * 16 + hi * 8 + v;
            int n = n0 + t * 16 + l;
            float val = acc[t][v] * oscale;
            if (MODE == 0) {
                int bb = m >> 11, s = m & (SEQ - 1);
                int hh = n >> 6,  dh = n & (DHEAD - 1);
                outBF[(size_t)((bb * NHEAD + hh) * SEQ + s) * DHEAD + dh] = f2bf(val);
            } else if (MODE == 2) {
                int bb = m >> 11, s = m & (SEQ - 1);
                int hh = n >> 6,  dh = n & (DHEAD - 1);
                int s64 = s & 63;
                int sp  = (s & ~63) + 4 * (s64 & 15) + (s64 >> 4);  // key-permuted slot
                outBF[(size_t)((bb * NHEAD + hh) * DHEAD + dh) * SEQ + sp] = f2bf(val);
            } else {
                outF[(size_t)m * D_MODEL + n] = val;
            }
        }
    }
}

// ---------------------------------------------------------------------------
// Flash attention: block = (qblock, head, batch), 4 waves x 16 query rows.
// Q,K: bf16 [B,H,S,64] (Q pre-scaled by log2e/8); V: bf16 [B,H,64,S'] permuted.
// ctx: fp32 [B,S,1024]. 64-key chunks; softmax in exp2 domain; row-sums via
// an extra WMMA against an all-ones B fragment.
// ---------------------------------------------------------------------------
__global__ __launch_bounds__(128) void attn_kernel(
    const unsigned short* __restrict__ Q,
    const unsigned short* __restrict__ K,
    const unsigned short* __restrict__ Vt,
    float* __restrict__ ctx)
{
    __shared__ __align__(16) unsigned short Kt[64][80];     // keys x dh   (160B rows)
    __shared__ __align__(16) unsigned short Vs[64][80];     // dh x keys'  (160B rows)
    __shared__ __align__(16) unsigned short Pb[4][16][80];  // per-wave P 16x64 (cols')

    const int tid  = threadIdx.x;
    const int wid  = tid >> 5;
    const int lane = tid & 31;
    const int l    = lane & 15;
    const int hi   = lane >> 4;

    const int qb = blockIdx.x;
    const int h  = blockIdx.y;
    const int b  = blockIdx.z;

    const unsigned short* Qh = Q  + (size_t)(b * NHEAD + h) * SEQ * DHEAD;
    const unsigned short* Kh = K  + (size_t)(b * NHEAD + h) * SEQ * DHEAD;
    const unsigned short* Vh = Vt + (size_t)(b * NHEAD + h) * DHEAD * SEQ;

    const int q0 = qb * 64 + wid * 16;   // this wave's 16 query rows
    const int k0 = hi ? 8 : 0;
    const int kb = hi ? 16 : 0;

    // Q A-fragments for dh chunks [0,32) and [32,64), loaded once.
    Frag16 aQ[2];
#pragma unroll
    for (int c = 0; c < 2; ++c) {
        aQ[c].q[0] = *(const uint4*)&Qh[(size_t)(q0 + l) * DHEAD + c * 32 + k0];
        aQ[c].q[1] = *(const uint4*)&Qh[(size_t)(q0 + l) * DHEAD + c * 32 + k0 + 16];
    }

    // Constant all-ones bf16 B fragment (rowsum via matrix engine).
    Frag16 ones;
    {
        uint4 o; o.x = o.y = o.z = o.w = 0x3F803F80u;   // bf16 1.0 pairs
        ones.q[0] = o; ones.q[1] = o;
    }

    v8f vacc[4];
#pragma unroll
    for (int t = 0; t < 4; ++t) vacc[t] = v8f_splat(0.0f);
    v8f mrow = v8f_splat(-3.0e38f);
    v8f lrow = v8f_splat(0.0f);

    for (int kk = 0; kk < SEQ; kk += 64) {
        // Cooperative staging: K row-major, V already transposed+permuted in global.
#pragma unroll
        for (int i = 0; i < 4; ++i) {
            int idx = i * 128 + tid;      // 0..511 chunks of 8 bf16
            int r   = idx >> 3;           // 0..63
            int c   = (idx & 7) * 8;      // 0..56
            *(uint4*)&Kt[r][c] = *(const uint4*)&Kh[(size_t)(kk + r) * DHEAD + c];
            *(uint4*)&Vs[r][c] = *(const uint4*)&Vh[(size_t)r * SEQ + kk + c];
        }
        // Warm L2 for the next chunk (global_prefetch_b8; speculative is safe).
        {
            int kn = (kk + 64 < SEQ) ? kk + 64 : kk;
            __builtin_prefetch(&Kh[(size_t)(kn + (tid & 63)) * DHEAD], 0, 0);
            __builtin_prefetch(&Vh[(size_t)(tid & 63) * SEQ + kn], 0, 0);
        }
        __syncthreads();

        // Scores: four 16-key tiles, S = Q(16x64) @ K^T(64x16) (scale pre-folded).
        v8f s[4];
#pragma unroll
        for (int sub = 0; sub < 4; ++sub) {
            v8f acc0 = v8f_splat(0.0f);
#pragma unroll
            for (int c = 0; c < 2; ++c) {
                Frag16 bk;  // B(kdim=dh, n=key) = Kt[key][dh]
                bk.q[0] = *(const uint4*)&Kt[sub * 16 + l][c * 32 + k0];
                bk.q[1] = *(const uint4*)&Kt[sub * 16 + l][c * 32 + k0 + 16];
                acc0 = __builtin_amdgcn_wmma_f32_16x16x32_bf16(
                    false, aQ[c].v, false, bk.v, (short)0, acc0, false, false);
            }
            s[sub] = acc0;
        }

        // Online softmax (exp2 domain). Lane's 4 P values for a row are packed
        // into one b64 store at column 4*l (key' = 4*l + sub matches V layout).
        v8f corr;
        uint2 pk[8];
#pragma unroll
        for (int v = 0; v < 8; ++v) {
            float x0 = s[0][v], x1 = s[1][v], x2 = s[2][v], x3 = s[3][v];
            float mx = fmaxf(fmaxf(x0, x1), fmaxf(x2, x3));
            mx = xmax16(mx);
            float mo = mrow[v];
            float mn = fmaxf(mo, mx);
            corr[v] = exp2f(mo - mn);
            mrow[v] = mn;
            float e0 = exp2f(x0 - mn);
            float e1 = exp2f(x1 - mn);
            float e2 = exp2f(x2 - mn);
            float e3 = exp2f(x3 - mn);
            pk[v].x = (unsigned)f2bf(e0) | ((unsigned)f2bf(e1) << 16);
            pk[v].y = (unsigned)f2bf(e2) | ((unsigned)f2bf(e3) << 16);
        }
#pragma unroll
        for (int t = 0; t < 4; ++t) vacc[t] *= corr;
#pragma unroll
        for (int v = 0; v < 8; ++v)
            *(uint2*)&Pb[wid][hi * 8 + v][4 * l] = pk[v];

        Frag16 aP[2];   // LDS ops in-order within a wave
#pragma unroll
        for (int g = 0; g < 2; ++g) {
            aP[g].q[0] = *(const uint4*)&Pb[wid][l][g * 32 + k0];
            aP[g].q[1] = *(const uint4*)&Pb[wid][l][g * 32 + k0 + 16];
        }

        // Row sums on the matrix engine: P(16x64) @ ones(64x16).
        v8f vsum = v8f_splat(0.0f);
        vsum = __builtin_amdgcn_wmma_f32_16x16x32_bf16(
            false, aP[0].v, false, ones.v, (short)0, vsum, false, false);
        vsum = __builtin_amdgcn_wmma_f32_16x16x32_bf16(
            false, aP[1].v, false, ones.v, (short)0, vsum, false, false);
        lrow = lrow * corr + vsum;

        // ctx += P(16x64) @ V(64x64); V B-frag: B(key', dh) = Vs[dh][key']
#pragma unroll
        for (int t = 0; t < 4; ++t) {
#pragma unroll
            for (int g = 0; g < 2; ++g) {
                Frag16 bv;
                bv.q[0] = *(const uint4*)&Vs[t * 16 + l][g * 32 + k0];
                bv.q[1] = *(const uint4*)&Vs[t * 16 + l][g * 32 + k0 + 16];
                vacc[t] = __builtin_amdgcn_wmma_f32_16x16x32_bf16(
                    false, aP[g].v, false, bv.v, (short)0, vacc[t], false, false);
            }
        }
        __syncthreads();
    }

    // Normalize and write context fp32 in [B,S,D] with head interleave.
#pragma unroll
    for (int t = 0; t < 4; ++t) {
#pragma unroll
        for (int v = 0; v < 8; ++v) {
            int q = q0 + hi * 8 + v;
            float val = vacc[t][v] / lrow[v];
            ctx[(size_t)(b * SEQ + q) * D_MODEL + h * DHEAD + t * 16 + l] = val;
        }
    }
}

// ---------------------------------------------------------------------------
// Workspace layout (needs 40 MB):
//   [0, 8MB)    Q bf16  [B,H,S,64]   (pre-scaled by log2e/8)
//   [8, 16MB)   K bf16  [B,H,S,64]
//   [16, 24MB)  V bf16  [B,H,64,S']  (transposed, key-permuted per 64-block)
//   [24, 40MB)  ctx fp32 [B,S,1024]
// ---------------------------------------------------------------------------
extern "C" void kernel_launch(void* const* d_in, const int* in_sizes, int n_in,
                              void* d_out, int out_size, void* d_ws, size_t ws_size,
                              hipStream_t stream) {
    (void)in_sizes; (void)n_in; (void)out_size; (void)ws_size;
    const float* x  = (const float*)d_in[0];
    const float* Wq = (const float*)d_in[1];
    const float* Wk = (const float*)d_in[2];
    const float* Wv = (const float*)d_in[3];
    const float* Wo = (const float*)d_in[4];

    unsigned short* qbf = (unsigned short*)d_ws;
    unsigned short* kbf = qbf + (size_t)MTOT * D_MODEL;
    unsigned short* vbf = kbf + (size_t)MTOT * D_MODEL;
    float*          ctx = (float*)(vbf + (size_t)MTOT * D_MODEL);

    const float qscale = 0.125f * 1.4426950408889634f;   // (1/sqrt(64)) * log2(e)

    dim3 gGemm(D_MODEL / 64, MTOT / 64);   // 16 x 64 blocks
    gemm_xWT<0><<<gGemm, 128, 0, stream>>>(x, Wq, qbf, nullptr, qscale);
    gemm_xWT<0><<<gGemm, 128, 0, stream>>>(x, Wk, kbf, nullptr, 1.0f);
    gemm_xWT<2><<<gGemm, 128, 0, stream>>>(x, Wv, vbf, nullptr, 1.0f);   // V transposed+permuted

    attn_kernel<<<dim3(SEQ / 64, NHEAD, BATCH), 128, 0, stream>>>(qbf, kbf, vbf, ctx);

    gemm_xWT<1><<<gGemm, 128, 0, stream>>>(ctx, Wo, nullptr, (float*)d_out, 1.0f);
}
